// CausalSelfAttention_13022340841799
// MI455X (gfx1250) — matrix operand order repaired and checked by reading, not compiled, not run on hardware
//
#include <hip/hip_runtime.h>
#include <hip/hip_bf16.h>

typedef __attribute__((ext_vector_type(16))) _Float16 v16h;
typedef __attribute__((ext_vector_type(8)))  float    v8f;
typedef __attribute__((ext_vector_type(4)))  int      v4i;

union H2 { unsigned int u; _Float16 h[2]; };

__device__ __forceinline__ int kpat(int r, int g) {
    return ((r < 4) ? 0 : 16) + 8 * g + 2 * (r & 3);
}

#define NEG_BIG (-3.0e38f)

// ---- gfx1250 async global->LDS copy (16B per lane), with sync fallback ----
__device__ __forceinline__ void async_cp16(_Float16* ldst, const _Float16* gsrc) {
#if __has_builtin(__builtin_amdgcn_global_load_async_to_lds_b128)
    __builtin_amdgcn_global_load_async_to_lds_b128(
        (__attribute__((address_space(1))) v4i*)const_cast<_Float16*>(gsrc),
        (__attribute__((address_space(3))) v4i*)ldst, 0, 0);
#else
    *(uint4*)ldst = *(const uint4*)gsrc;
#endif
}

__device__ __forceinline__ void wait_async0() {
#if __has_builtin(__builtin_amdgcn_s_wait_asynccnt)
    __builtin_amdgcn_s_wait_asynccnt(0);
#else
    asm volatile("s_wait_asynccnt 0" ::: "memory");
#endif
}

// ---------------------------------------------------------------------------
// Prep kernels (run once, tiny): fp32 -> f16 staging so the GEMM hot loops
// are pure 16B async copies.
// ---------------------------------------------------------------------------
__global__ __launch_bounds__(256)
void cvt_f32_to_f16_kernel(const float* __restrict__ in, _Float16* __restrict__ out,
                           int n4) {
    int i = blockIdx.x * 256 + threadIdx.x;
    if (i < n4) {
        float4 f = ((const float4*)in)[i];
        _Float16* d = out + (size_t)i * 4;
        d[0] = (_Float16)f.x; d[1] = (_Float16)f.y;
        d[2] = (_Float16)f.z; d[3] = (_Float16)f.w;
    }
}

// W[k][n] fp32 (K=512 rows) -> WT[n][k] f16 (k-contiguous)
__global__ __launch_bounds__(256)
void cvt_transpose_kernel(const float* __restrict__ W, _Float16* __restrict__ WT,
                          int N) {
    int gid = blockIdx.x * 256 + threadIdx.x;   // n*128 + k4
    int n  = gid >> 7;
    int k4 = gid & 127;
    if (n < N) {
        _Float16* d = WT + (size_t)n * 512 + k4 * 4;
#pragma unroll
        for (int j = 0; j < 4; ++j)
            d[j] = (_Float16)W[(size_t)(k4 * 4 + j) * N + n];
    }
}

// ---------------------------------------------------------------------------
// Shared WMMA GEMM: C[M,N] = A[M,512] (f16, row-major) x BT[N,512] (f16,
// n-major/k-contig) + bias.  BM=128 BN=128 BK=32, 8 waves, double-buffered
// LDS fed by async global->LDS copies.  Row stride padded to 40 halves
// (bank stride 20 -> conflict-free b128 fragment loads).
// ---------------------------------------------------------------------------
#define GSTRIDE 40

template<bool QKV>
__global__ __launch_bounds__(256)
void gemm_kernel(const _Float16* __restrict__ A, const _Float16* __restrict__ BT,
                 const float* __restrict__ bias,
                 _Float16* __restrict__ Q, _Float16* __restrict__ K,
                 _Float16* __restrict__ V, float* __restrict__ Out) {
    __shared__ __align__(16) _Float16 As[2][128 * GSTRIDE];
    __shared__ __align__(16) _Float16 Bs[2][128 * GSTRIDE];

    const int tid  = threadIdx.x;
    const int lane = tid & 31;
    const int w    = tid >> 5;
    const int wm   = w & 3;
    const int wn   = w >> 2;
    const int g    = lane >> 4;
    const int ln   = lane & 15;

    const int m0 = blockIdx.y * 128;
    const int n0 = blockIdx.x * 128;

    v8f acc[2][4];
#pragma unroll
    for (int i = 0; i < 2; ++i)
#pragma unroll
        for (int j = 0; j < 4; ++j)
#pragma unroll
            for (int r = 0; r < 8; ++r) acc[i][j][r] = 0.0f;

    // 512 chunks of 8 halves per 128x32 tile; each thread owns 2 chunks.
    auto copy_tiles = [&](int buf, int k0) {
#pragma unroll
        for (int i = 0; i < 2; ++i) {
            int c   = tid + 256 * i;
            int row = c >> 2;
            int c8  = c & 3;
            async_cp16(&As[buf][row * GSTRIDE + c8 * 8],
                       A  + (size_t)(m0 + row) * 512 + k0 + c8 * 8);
            async_cp16(&Bs[buf][row * GSTRIDE + c8 * 8],
                       BT + (size_t)(n0 + row) * 512 + k0 + c8 * 8);
        }
    };

    copy_tiles(0, 0);
    wait_async0();
    __syncthreads();

    for (int kt = 0; kt < 16; ++kt) {
        const int cur = kt & 1;
        if (kt + 1 < 16) copy_tiles(cur ^ 1, (kt + 1) * 32);

        const _Float16* Asc = As[cur];
        const _Float16* Bsc = Bs[cur];

        v16h afr[2], bfr[4];
#pragma unroll
        for (int i = 0; i < 2; ++i) {
            int m = wm * 32 + i * 16 + ln;
#pragma unroll
            for (int r = 0; r < 8; ++r) {
                H2 c; c.u = *(const unsigned int*)(&Asc[m * GSTRIDE + kpat(r, g)]);
                afr[i][2 * r] = c.h[0]; afr[i][2 * r + 1] = c.h[1];
            }
        }
#pragma unroll
        for (int j = 0; j < 4; ++j) {
            int n = wn * 64 + j * 16 + ln;
#pragma unroll
            for (int r = 0; r < 8; ++r) {
                H2 c; c.u = *(const unsigned int*)(&Bsc[n * GSTRIDE + kpat(r, g)]);
                bfr[j][2 * r] = c.h[0]; bfr[j][2 * r + 1] = c.h[1];
            }
        }
#pragma unroll
        for (int i = 0; i < 2; ++i)
#pragma unroll
            for (int j = 0; j < 4; ++j)
                acc[i][j] = __builtin_amdgcn_wmma_f32_16x16x32_f16(
                    false, afr[i], false, bfr[j], (short)0, acc[i][j], false, false);

        wait_async0();
        __syncthreads();
    }

#pragma unroll
    for (int i = 0; i < 2; ++i)
#pragma unroll
        for (int j = 0; j < 4; ++j)
#pragma unroll
            for (int r = 0; r < 8; ++r) {
                int mg = m0 + wm * 32 + i * 16 + r + 8 * g;
                int ng = n0 + wn * 64 + j * 16 + ln;
                float val = acc[i][j][r] + bias[ng];
                if (QKV) {
                    int which = ng >> 9;          // 0=Q 1=K 2=V
                    int col   = ng & 511;
                    int h     = col >> 6;
                    int dc    = col & 63;
                    int bb    = mg >> 12;
                    int pos   = mg & 4095;
                    size_t idx = (((size_t)(bb * 8 + h)) * 4096 + pos) * 64 + dc;
                    _Float16* dst = (which == 0) ? Q : (which == 1) ? K : V;
                    dst[idx] = (_Float16)val;
                } else {
                    Out[(size_t)mg * 512 + ng] = val;
                }
            }
}

// ---------------------------------------------------------------------------
// Fused flash attention per (b,h,query-tile-of-64).  4 waves x 16 query rows.
// LDS row strides padded to 72 halves (bank stride 36 -> conflict-free).
// ---------------------------------------------------------------------------
#define FSTRIDE 72

__global__ __launch_bounds__(128)
void flash_attn_kernel(const _Float16* __restrict__ Q,
                       const _Float16* __restrict__ K,
                       const _Float16* __restrict__ V,
                       _Float16* __restrict__ O) {
    __shared__ __align__(16) _Float16 Ks[64 * FSTRIDE];       // [key][d]
    __shared__ __align__(16) _Float16 Vt[64 * FSTRIDE];       // [d][key]
    __shared__ __align__(16) _Float16 Ps[4][16 * FSTRIDE];    // per-wave P

    const int bidx = blockIdx.x;
    const int qt   = bidx & 63;
    const int bh   = bidx >> 6;
    const _Float16* Qh = Q + (size_t)bh * 4096 * 64;
    const _Float16* Kh = K + (size_t)bh * 4096 * 64;
    const _Float16* Vh = V + (size_t)bh * 4096 * 64;

    const int tid  = threadIdx.x;
    const int lane = tid & 31;
    const int w    = tid >> 5;
    const int g    = lane >> 4;
    const int ln   = lane & 15;
    const int q0   = qt * 64;
    const int qrow = q0 + w * 16;

    v16h qfr[2];
#pragma unroll
    for (int kk = 0; kk < 2; ++kk)
#pragma unroll
        for (int r = 0; r < 8; ++r) {
            int d = 32 * kk + kpat(r, g);
            H2 c; c.u = *(const unsigned int*)(Qh + (size_t)(qrow + ln) * 64 + d);
            qfr[kk][2 * r] = c.h[0]; qfr[kk][2 * r + 1] = c.h[1];
        }

    v8f acc[4];
    float rm[8], rs[8];
#pragma unroll
    for (int nt = 0; nt < 4; ++nt)
#pragma unroll
        for (int r = 0; r < 8; ++r) acc[nt][r] = 0.0f;
#pragma unroll
    for (int r = 0; r < 8; ++r) { rm[r] = NEG_BIG; rs[r] = 0.0f; }

    for (int kb0 = 0; kb0 <= q0; kb0 += 64) {
        // K tile: pure copy -> async DMA into LDS
#pragma unroll
        for (int i = 0; i < 4; ++i) {
            int li = tid + 128 * i;
            int row = li >> 3, c8 = li & 7;
            async_cp16(&Ks[row * FSTRIDE + c8 * 8],
                       Kh + (size_t)(kb0 + row) * 64 + c8 * 8);
        }
        // V tile transposed scatter (sync)
#pragma unroll
        for (int i = 0; i < 4; ++i) {
            int li = tid + 128 * i;
            int key = li >> 3, d8 = li & 7;
            uint4 f = *(const uint4*)(Vh + (size_t)(kb0 + key) * 64 + d8 * 8);
            const _Float16* hp = (const _Float16*)&f;
#pragma unroll
            for (int j2 = 0; j2 < 8; ++j2) Vt[(d8 * 8 + j2) * FSTRIDE + key] = hp[j2];
        }
        wait_async0();
        __syncthreads();

        // S = Q K^T
        v8f s[4];
#pragma unroll
        for (int n = 0; n < 4; ++n) {
            v8f c;
#pragma unroll
            for (int r = 0; r < 8; ++r) c[r] = 0.0f;
#pragma unroll
            for (int kk = 0; kk < 2; ++kk) {
                v16h bfr;
                int kl = n * 16 + ln;
#pragma unroll
                for (int r = 0; r < 8; ++r) {
                    int kd = 32 * kk + kpat(r, g);
                    H2 cc; cc.u = *(const unsigned int*)(&Ks[kl * FSTRIDE + kd]);
                    bfr[2 * r] = cc.h[0]; bfr[2 * r + 1] = cc.h[1];
                }
                c = __builtin_amdgcn_wmma_f32_16x16x32_f16(
                        false, qfr[kk], false, bfr, (short)0, c, false, false);
            }
            s[n] = c;
        }

        // Scale + mask (causal + prefix=1) on boundary/first blocks only
        const bool needmask = (kb0 + 63 > qrow) || (kb0 == 0);
#pragma unroll
        for (int n = 0; n < 4; ++n)
#pragma unroll
            for (int r = 0; r < 8; ++r) {
                float sv = s[n][r] * 0.125f;
                if (needmask) {
                    int qg = qrow + r + 8 * g;
                    int kg = kb0 + n * 16 + ln;
                    bool ok = (kg <= qg) && ((qg < 1) || (kg >= 1));
                    sv = ok ? sv : NEG_BIG;
                }
                s[n][r] = sv;
            }

        // Online softmax
        float alpha[8];
#pragma unroll
        for (int r = 0; r < 8; ++r) {
            float lm = s[0][r];
#pragma unroll
            for (int n = 1; n < 4; ++n) lm = fmaxf(lm, s[n][r]);
#pragma unroll
            for (int off = 1; off < 16; off <<= 1)
                lm = fmaxf(lm, __shfl_xor(lm, off, 32));
            float mo = rm[r];
            float mn = fmaxf(mo, lm);
            alpha[r] = __expf(mo - mn);
            rm[r] = mn;
        }
#pragma unroll
        for (int n = 0; n < 4; ++n)
#pragma unroll
            for (int r = 0; r < 8; ++r) s[n][r] = __expf(s[n][r] - rm[r]);
#pragma unroll
        for (int r = 0; r < 8; ++r) {
            float sm = s[0][r] + s[1][r] + s[2][r] + s[3][r];
#pragma unroll
            for (int off = 1; off < 16; off <<= 1)
                sm += __shfl_xor(sm, off, 32);
            rs[r] = rs[r] * alpha[r] + sm;
        }
#pragma unroll
        for (int nt = 0; nt < 4; ++nt)
#pragma unroll
            for (int r = 0; r < 8; ++r) acc[nt][r] *= alpha[r];

        // C-layout -> A-fragment layout via per-wave LDS staging
        _Float16* Pw = &Ps[w][0];
#pragma unroll
        for (int n = 0; n < 4; ++n)
#pragma unroll
            for (int r = 0; r < 8; ++r) {
                int m = r + 8 * g;
                Pw[m * FSTRIDE + n * 16 + ln] = (_Float16)s[n][r];
            }

        // O += P x V
#pragma unroll
        for (int nt = 0; nt < 4; ++nt) {
#pragma unroll
            for (int kk = 0; kk < 2; ++kk) {
                v16h afr, bfr;
#pragma unroll
                for (int r = 0; r < 8; ++r) {
                    int kl = 32 * kk + kpat(r, g);
                    H2 cc; cc.u = *(const unsigned int*)(&Pw[ln * FSTRIDE + kl]);
                    afr[2 * r] = cc.h[0]; afr[2 * r + 1] = cc.h[1];
                }
                int dl = nt * 16 + ln;
#pragma unroll
                for (int r = 0; r < 8; ++r) {
                    int key = 32 * kk + kpat(r, g);
                    H2 cc; cc.u = *(const unsigned int*)(&Vt[dl * FSTRIDE + key]);
                    bfr[2 * r] = cc.h[0]; bfr[2 * r + 1] = cc.h[1];
                }
                acc[nt] = __builtin_amdgcn_wmma_f32_16x16x32_f16(
                            false, afr, false, bfr, (short)0, acc[nt], false, false);
            }
        }
        __syncthreads();
    }

    const int b = bh >> 3;
    const int h = bh & 7;
#pragma unroll
    for (int nt = 0; nt < 4; ++nt)
#pragma unroll
        for (int r = 0; r < 8; ++r) {
            int qg = qrow + r + 8 * g;
            int dg = h * 64 + nt * 16 + ln;
            float val = acc[nt][r] / rs[r];
            O[(size_t)(b * 4096 + qg) * 512 + dg] = (_Float16)val;
        }
}

// ---------------------------------------------------------------------------
extern "C" void kernel_launch(void* const* d_in, const int* in_sizes, int n_in,
                              void* d_out, int out_size, void* d_ws, size_t ws_size,
                              hipStream_t stream) {
    const float* x     = (const float*)d_in[0];
    const float* w_in  = (const float*)d_in[1];
    const float* b_in  = (const float*)d_in[2];
    const float* w_out = (const float*)d_in[3];
    const float* b_out = (const float*)d_in[4];

    _Float16* ws = (_Float16*)d_ws;
    const size_t headElems = (size_t)2 * 8 * 4096 * 64;   // 4,194,304
    _Float16* Qb    = ws;
    _Float16* Kb    = ws + headElems;
    _Float16* Vb    = ws + 2 * headElems;
    _Float16* Ab    = ws + 3 * headElems;                 // attn out [8192,512]
    _Float16* Xh    = ws + 4 * headElems;                 // x as f16 [8192,512]
    _Float16* WinT  = Xh + (size_t)8192 * 512;            // [1536][512]
    _Float16* WoutT = WinT + (size_t)1536 * 512;          // [512][512]

    // 0) one-shot f16 staging (pure-copy GEMM hot loops)
    cvt_f32_to_f16_kernel<<<4096, 256, 0, stream>>>(x, Xh, 8192 * 512 / 4);
    cvt_transpose_kernel<<<768, 256, 0, stream>>>(w_in, WinT, 1536);
    cvt_transpose_kernel<<<256, 256, 0, stream>>>(w_out, WoutT, 512);

    // 1) QKV projection
    gemm_kernel<true><<<dim3(12, 64), 256, 0, stream>>>(
        Xh, WinT, b_in, Qb, Kb, Vb, nullptr);

    // 2) Fused flash attention
    flash_attn_kernel<<<dim3(1024), 128, 0, stream>>>(Qb, Kb, Vb, Ab);

    // 3) Output projection
    gemm_kernel<false><<<dim3(4, 64), 256, 0, stream>>>(
        Ab, WoutT, b_out, nullptr, nullptr, nullptr, (float*)d_out);
}
